// MODCANGNNModel_72301479461384
// MI455X (gfx1250) — compile-verified, matrix operand order checked by reading, not run on hardware
//
#include <hip/hip_runtime.h>
#include <hip/hip_bf16.h>

// MODCAN GNN forward for MI455X (gfx1250, wave32).
// Dense layers: bf16 WMMA (v_wmma_f32_16x16x32_bf16), f32 accumulate.
// Edge phase: 3-pass segment softmax + scatter with hw f32/u32 atomics (L2 resident).

typedef __attribute__((ext_vector_type(16))) __bf16   v16bf;
typedef __attribute__((ext_vector_type(8)))  float    v8f;
typedef __attribute__((ext_vector_type(8)))  unsigned v8u;

// pack two f32 -> two bf16 (RNE) in one dword: lo in [15:0], hi in [31:16]
__device__ __forceinline__ unsigned pack2bf(float lo, float hi) {
  unsigned ul = __builtin_bit_cast(unsigned, lo);
  ul += 0x7FFFu + ((ul >> 16) & 1u);
  unsigned uh = __builtin_bit_cast(unsigned, hi);
  uh += 0x7FFFu + ((uh >> 16) & 1u);
  return (ul >> 16) | (uh & 0xFFFF0000u);
}

// Ordered-uint encoding so atomicMax(u32) == float max
__device__ __forceinline__ unsigned enc_f32(float f) {
  unsigned u = __builtin_bit_cast(unsigned, f);
  return (u & 0x80000000u) ? ~u : (u | 0x80000000u);
}
__device__ __forceinline__ float dec_f32(unsigned u) {
  unsigned b = (u & 0x80000000u) ? (u ^ 0x80000000u) : ~u;
  return __builtin_bit_cast(float, b);
}
#define ENC_NEG_INF 0x007FFFFFu  // enc(-inf)

// ---------------------------------------------------------------------------
// GEMM: C[N,M] = A[N,K] @ W[K,M], f32 in/out, bf16 WMMA compute.
// Block = 128 threads (4 waves). Block tile 64x64, K step 32.
// Wave w computes rows [w*16, w*16+16) x all 64 cols (4 wmma accumulators).
// LDS tiles hold packed bf16 pairs (dwords), row stride 17 dwords (conflict-free).
// ---------------------------------------------------------------------------
#define LDS_STRIDE 17

__global__ __launch_bounds__(128) void gemm_bf16_wmma(
    const float* __restrict__ A, const float* __restrict__ W,
    float* __restrict__ C, int Nrows, int K, int M) {
  __shared__ unsigned lA[64 * LDS_STRIDE];   // A tile: 64 rows x 32 k (bf16 pairs)
  __shared__ unsigned lBT[64 * LDS_STRIDE];  // W tile transposed: 64 cols x 32 k
  const int tid  = threadIdx.x;
  const int lane = tid & 31;
  const int wid  = tid >> 5;
  const int half = lane >> 4;   // lane group 0..15 vs 16..31
  const int rl   = lane & 15;
  const int rBase = blockIdx.x * 64;
  const int cBase = blockIdx.y * 64;

  const v8f vz = {0.f, 0.f, 0.f, 0.f, 0.f, 0.f, 0.f, 0.f};
  v8f acc[4];
  acc[0] = vz; acc[1] = vz; acc[2] = vz; acc[3] = vz;

  const int aRow = (wid * 16 + rl) * LDS_STRIDE;  // this lane's A-fragment row

  for (int k0 = 0; k0 < K; k0 += 32) {
    // ---- A tile 64x32: float2 per thread, coalesced; clamp rows (stores are
    // guarded later, so duplicated padding rows are harmless) ----
    #pragma unroll
    for (int i = 0; i < 8; ++i) {
      int idx = tid + i * 128;                 // 0..1023 float2 slots
      int r = idx >> 4, c2 = (idx & 15) << 1;  // row, even col
      int gr = rBase + r;
      if (gr >= Nrows) gr = Nrows - 1;
      const float2 v = *(const float2*)&A[(long)gr * K + (k0 + c2)];
      lA[r * LDS_STRIDE + (c2 >> 1)] = pack2bf(v.x, v.y);
    }
    // ---- W tile 32x64, stored transposed: 2x2 patches, float2 loads along
    // the contiguous column axis, dword LDS stores ----
    #pragma unroll
    for (int i = 0; i < 4; ++i) {
      int pi = tid + i * 128;                  // 0..511 patches
      int pr = (pi >> 5) << 1;                 // even k row
      int pc = (pi & 31) << 1;                 // even col
      const float2 v0 = *(const float2*)&W[(long)(k0 + pr) * M + (cBase + pc)];
      const float2 v1 = *(const float2*)&W[(long)(k0 + pr + 1) * M + (cBase + pc)];
      lBT[pc * LDS_STRIDE + (pr >> 1)]       = pack2bf(v0.x, v1.x);
      lBT[(pc + 1) * LDS_STRIDE + (pr >> 1)] = pack2bf(v0.y, v1.y);
    }
    __syncthreads();

    // A fragment (documented 16-bit 16x32 layout), assembled as 8 dwords:
    // dword j<4: k-pair index j + 4*half ; j>=4: j + 4 + 4*half
    v8u au;
    #pragma unroll
    for (int j = 0; j < 8; ++j) {
      int kd = (j < 4) ? (j + 4 * half) : (j + 4 + 4 * half);
      au[j] = lA[aRow + kd];
    }
    const v16bf a = __builtin_bit_cast(v16bf, au);

    #pragma unroll
    for (int ct = 0; ct < 4; ++ct) {
      // B fragment: transpose-of-A layout (lane holds column n = rl)
      v8u bu;
      #pragma unroll
      for (int j = 0; j < 8; ++j) {
        int kd = (j < 4) ? (j + 4 * half) : (j + 4 + 4 * half);
        bu[j] = lBT[(ct * 16 + rl) * LDS_STRIDE + kd];
      }
      const v16bf b = __builtin_bit_cast(v16bf, bu);
      acc[ct] = __builtin_amdgcn_wmma_f32_16x16x32_bf16(
          false, a, false, b, (short)0, acc[ct], false, false);
    }
    __syncthreads();
  }

  // C/D layout: lane<16 holds rows 0..7, lane>=16 rows 8..15; col = rl
  #pragma unroll
  for (int ct = 0; ct < 4; ++ct) {
    #pragma unroll
    for (int i = 0; i < 8; ++i) {
      int gr = rBase + wid * 16 + i + 8 * half;
      if (gr < Nrows)
        C[(long)gr * M + (cBase + ct * 16 + rl)] = acc[ct][i];
    }
  }
}

// ---------------------------------------------------------------------------
// LayerNorm(x + bias) * g + b, optional ReLU. One block (F threads) per node.
// ---------------------------------------------------------------------------
__global__ void ln_act_kernel(const float* __restrict__ in, int inStride,
                              const float* __restrict__ bias,
                              const float* __restrict__ g,
                              const float* __restrict__ b,
                              float* __restrict__ out, int outStride, int outOff,
                              int act) {
  const int F = blockDim.x;
  const int n = blockIdx.x, c = threadIdx.x;
  __shared__ float red[256];
  float x = in[(long)n * inStride + c] + bias[c];
  red[c] = x; __syncthreads();
  for (int s = F >> 1; s > 0; s >>= 1) {
    if (c < s) red[c] += red[c + s];
    __syncthreads();
  }
  float m = red[0] / (float)F; __syncthreads();
  float dx = x - m;
  red[c] = dx * dx; __syncthreads();
  for (int s = F >> 1; s > 0; s >>= 1) {
    if (c < s) red[c] += red[c + s];
    __syncthreads();
  }
  float v = red[0] / (float)F;
  float y = dx * rsqrtf(v + 1e-5f) * g[c] + b[c];
  if (act == 1) y = fmaxf(y, 0.f);
  out[(long)n * outStride + outOff + c] = y;
}

// a_src[n,h] = sum_d xw[n,h*64+d]*att_src[h,d]; likewise a_dst. 256 thr/node.
__global__ void att_kernel(const float* __restrict__ xw,
                           const float* __restrict__ as,
                           const float* __restrict__ ad,
                           float* __restrict__ a_src, float* __restrict__ a_dst) {
  const int n = blockIdx.x, c = threadIdx.x;
  __shared__ float s1[256], s2[256];
  float x = xw[(long)n * 256 + c];
  s1[c] = x * as[c];
  s2[c] = x * ad[c];
  __syncthreads();
  for (int s = 32; s > 0; s >>= 1) {
    if ((c & 63) < s) { s1[c] += s1[c + s]; s2[c] += s2[c + s]; }
    __syncthreads();
  }
  if ((c & 63) == 0) {
    a_src[n * 4 + (c >> 6)] = s1[c];
    a_dst[n * 4 + (c >> 6)] = s2[c];
  }
}

__global__ void fill_u32(unsigned* __restrict__ p, unsigned v, long count) {
  long i = (long)blockIdx.x * blockDim.x + threadIdx.x;
  if (i < count) p[i] = v;
}

__device__ __forceinline__ void edge_sd(const int* ei, int E, int e, int& s, int& d) {
  if (e < E) { s = ei[e]; d = ei[E + e]; }
  else       { s = e - E; d = e - E; }   // self loop
}

// Pass 1: segment max over dst (per head)
__global__ void edge_max_kernel(const int* __restrict__ ei, int E, int Nn,
                                const float* __restrict__ a_src,
                                const float* __restrict__ a_dst,
                                unsigned* __restrict__ mx) {
  long i = (long)blockIdx.x * blockDim.x + threadIdx.x;
  long total = (long)(E + Nn) * 4;
  if (i >= total) return;
  int e = (int)(i >> 2), h = (int)(i & 3);
  int s, d; edge_sd(ei, E, e, s, d);
  float v = a_src[s * 4 + h] + a_dst[d * 4 + h];
  v = (v >= 0.f) ? v : 0.2f * v;            // LeakyReLU(0.2)
  atomicMax(&mx[d * 4 + h], enc_f32(v));
}

// Pass 2: denominator sum of exp
__global__ void edge_den_kernel(const int* __restrict__ ei, int E, int Nn,
                                const float* __restrict__ a_src,
                                const float* __restrict__ a_dst,
                                const unsigned* __restrict__ mx,
                                float* __restrict__ den) {
  long i = (long)blockIdx.x * blockDim.x + threadIdx.x;
  long total = (long)(E + Nn) * 4;
  if (i >= total) return;
  int e = (int)(i >> 2), h = (int)(i & 3);
  int s, d; edge_sd(ei, E, e, s, d);
  float v = a_src[s * 4 + h] + a_dst[d * 4 + h];
  v = (v >= 0.f) ? v : 0.2f * v;
  unsafeAtomicAdd(&den[d * 4 + h], __expf(v - dec_f32(mx[d * 4 + h])));
}

// Pass 3: weighted scatter of messages. One 256-thread block per edge.
__global__ void edge_msg_kernel(const int* __restrict__ ei, int E, int Nn,
                                const float* __restrict__ a_src,
                                const float* __restrict__ a_dst,
                                const unsigned* __restrict__ mx,
                                const float* __restrict__ den,
                                const float* __restrict__ xw,
                                float* __restrict__ hc) {
  const int e = blockIdx.x, c = threadIdx.x;
  int s, d; edge_sd(ei, E, e, s, d);
  __shared__ float alpha[4];
  if (c < 4) {
    float v = a_src[s * 4 + c] + a_dst[d * 4 + c];
    v = (v >= 0.f) ? v : 0.2f * v;
    alpha[c] = __expf(v - dec_f32(mx[d * 4 + c])) / den[d * 4 + c];
  }
  __syncthreads();
  unsafeAtomicAdd(&hc[(long)d * 256 + c], xw[(long)s * 256 + c] * alpha[c >> 6]);
}

// h = ELU(LN(0.8*(hc+bias) + 0.2*h0)). 256 threads per node.
__global__ void gat_epilogue_kernel(const float* __restrict__ hc,
                                    const float* __restrict__ bias,
                                    const float* __restrict__ h0,
                                    const float* __restrict__ g,
                                    const float* __restrict__ b,
                                    float* __restrict__ h) {
  const int n = blockIdx.x, c = threadIdx.x;
  __shared__ float red[256];
  float x = 0.8f * (hc[(long)n * 256 + c] + bias[c]) + 0.2f * h0[(long)n * 256 + c];
  red[c] = x; __syncthreads();
  for (int s = 128; s > 0; s >>= 1) { if (c < s) red[c] += red[c + s]; __syncthreads(); }
  float m = red[0] * (1.f / 256.f); __syncthreads();
  float dx = x - m;
  red[c] = dx * dx; __syncthreads();
  for (int s = 128; s > 0; s >>= 1) { if (c < s) red[c] += red[c + s]; __syncthreads(); }
  float v = red[0] * (1.f / 256.f);
  float y = dx * rsqrtf(v + 1e-5f) * g[c] + b[c];
  y = (y > 0.f) ? y : (__expf(y) - 1.f);    // ELU
  h[(long)n * 256 + c] = y;
}

// logits[n] = dot(z[n,:], w2) + b2. 128 threads per node.
__global__ void logits_kernel(const float* __restrict__ z,
                              const float* __restrict__ w2,
                              const float* __restrict__ b2,
                              float* __restrict__ out) {
  const int n = blockIdx.x, c = threadIdx.x;
  __shared__ float red[128];
  red[c] = z[(long)n * 128 + c] * w2[c];
  __syncthreads();
  for (int s = 64; s > 0; s >>= 1) { if (c < s) red[c] += red[c + s]; __syncthreads(); }
  if (c == 0) out[n] = red[0] + b2[0];
}

// ---------------------------------------------------------------------------
extern "C" void kernel_launch(void* const* d_in, const int* in_sizes, int n_in,
                              void* d_out, int out_size, void* d_ws, size_t ws_size,
                              hipStream_t stream) {
  const int N = in_sizes[0] / 128;   // 50000
  const int E = in_sizes[1] / 2;     // 1600000
  const int EE = E + N;              // edges + self loops

  auto F = [&](int i) { return (const float*)d_in[i]; };
  const float* omics = F(0);
  const int*   ei    = (const int*)d_in[1];
  const float* topo  = F(2);
  // params in setup_inputs() insertion order:
  const float* om_w1 = F(3),  *om_b1 = F(4),  *om_g1 = F(5),  *om_bn1 = F(6);
  const float* om_w2 = F(7),  *om_b2 = F(8),  *om_g2 = F(9),  *om_bn2 = F(10);
  const float* tp_w  = F(11), *tp_b  = F(12), *tp_g  = F(13), *tp_bn  = F(14);
  // gnn layer params at base 15 and 21: W, att_src, att_dst, bias, ln_g, ln_b
  const float* cl_w1 = F(27), *cl_b1 = F(28), *cl_g = F(29), *cl_bn = F(30);
  const float* cl_w2 = F(31), *cl_b2 = F(32);

  // workspace carve-up
  char* ws = (char*)d_ws;
  size_t big = (size_t)N * 256 * sizeof(float);
  float*    h0    = (float*)ws;            ws += big;
  float*    h     = (float*)ws;            ws += big;
  float*    xw    = (float*)ws;            ws += big;
  float*    hc    = (float*)ws;            ws += big;
  float*    a_src = (float*)ws;            ws += (size_t)N * 4 * sizeof(float);
  float*    a_dst = (float*)ws;            ws += (size_t)N * 4 * sizeof(float);
  unsigned* mx    = (unsigned*)ws;         ws += (size_t)N * 4 * sizeof(unsigned);
  float*    den   = (float*)ws;            ws += (size_t)N * 4 * sizeof(float);
  if ((size_t)(ws - (char*)d_ws) > ws_size) return;  // insufficient scratch

  auto gGrid = [&](int M) { return dim3((unsigned)((N + 63) / 64), (unsigned)(M / 64)); };
  const long eTot = (long)EE * 4;
  const unsigned eBlocks = (unsigned)((eTot + 255) / 256);

  // ---- encoder ----
  gemm_bf16_wmma<<<gGrid(256), 128, 0, stream>>>(omics, om_w1, xw, N, 128, 256);
  ln_act_kernel<<<N, 256, 0, stream>>>(xw, 256, om_b1, om_g1, om_bn1, hc, 256, 0, 1);
  gemm_bf16_wmma<<<gGrid(128), 128, 0, stream>>>(hc, om_w2, xw, N, 256, 128);
  ln_act_kernel<<<N, 128, 0, stream>>>(xw, 128, om_b2, om_g2, om_bn2, h0, 256, 0, 0);
  gemm_bf16_wmma<<<gGrid(128), 128, 0, stream>>>(topo, tp_w, xw, N, 128, 128);
  ln_act_kernel<<<N, 128, 0, stream>>>(xw, 128, tp_b, tp_g, tp_bn, h0, 256, 128, 1);

  // ---- 2 GAT layers ----
  const float* hin = h0;
  for (int L = 0; L < 2; ++L) {
    int p = 15 + L * 6;
    const float* W    = F(p);
    const float* ats  = F(p + 1);
    const float* atd  = F(p + 2);
    const float* bias = F(p + 3);
    const float* lng  = F(p + 4);
    const float* lnb  = F(p + 5);

    gemm_bf16_wmma<<<gGrid(256), 128, 0, stream>>>(hin, W, xw, N, 256, 256);
    att_kernel<<<N, 256, 0, stream>>>(xw, ats, atd, a_src, a_dst);

    fill_u32<<<(N * 4 + 255) / 256, 256, 0, stream>>>(mx, ENC_NEG_INF, (long)N * 4);
    fill_u32<<<(N * 4 + 255) / 256, 256, 0, stream>>>((unsigned*)den, 0u, (long)N * 4);
    fill_u32<<<((long)N * 256 + 255) / 256, 256, 0, stream>>>((unsigned*)hc, 0u, (long)N * 256);

    edge_max_kernel<<<eBlocks, 256, 0, stream>>>(ei, E, N, a_src, a_dst, mx);
    edge_den_kernel<<<eBlocks, 256, 0, stream>>>(ei, E, N, a_src, a_dst, mx, den);
    edge_msg_kernel<<<EE, 256, 0, stream>>>(ei, E, N, a_src, a_dst, mx, den, xw, hc);

    gat_epilogue_kernel<<<N, 256, 0, stream>>>(hc, bias, h0, lng, lnb, h);
    hin = h;
  }

  // ---- classifier ----
  gemm_bf16_wmma<<<gGrid(128), 128, 0, stream>>>(h, cl_w1, xw, N, 256, 128);
  ln_act_kernel<<<N, 128, 0, stream>>>(xw, 128, cl_b1, cl_g, cl_bn, hc, 128, 0, 1);
  logits_kernel<<<N, 128, 0, stream>>>(hc, cl_w2, cl_b2, (float*)d_out);
}
